// Ours_30468497998545
// MI455X (gfx1250) — matrix operand order repaired and checked by reading, not compiled, not run on hardware
//
#include <hip/hip_runtime.h>
#include <hip/hip_bf16.h>
#include <stdint.h>

typedef __attribute__((ext_vector_type(2)))  float    v2f;
typedef __attribute__((ext_vector_type(4)))  float    v4f;
typedef __attribute__((ext_vector_type(8)))  float    v8f;
typedef __attribute__((ext_vector_type(16))) _Float16 v16h;

#define N_NODES 4096
#define IN_F    256
#define F_OUT   64
#define H_HEADS 4
#define NEG_SLOPE 0.2f

__device__ __forceinline__ float leaky(float v) { return v > 0.f ? v : NEG_SLOPE * v; }

// ---------------------------------------------------------------------------
// Kernel 1: h = x @ W  (4096x256 * 256x256), exact fp32 via v_wmma_f32_16x16x4_f32.
// Also emits hT in f16, layout hT[f_global][node] for the attention kernel's
// B-operand (contiguous K per lane). One wave per 16x16 output tile.
// ---------------------------------------------------------------------------
__global__ __launch_bounds__(128) void k_gemm_xw(const float* __restrict__ x,
                                                 const float* __restrict__ W,
                                                 float* __restrict__ h,
                                                 _Float16* __restrict__ hT)
{
    const int wave = threadIdx.x >> 5;
    const int lane = threadIdx.x & 31;
    const int tile = blockIdx.x * 4 + wave;      // 4096 tiles = 256 mtiles * 16 ntiles
    const int mtile = tile >> 4;
    const int ntile = tile & 15;
    const int row  = mtile * 16 + (lane & 15);   // A: lanes 0-15 -> M, lanes 16-31 same M
    const int ncol = ntile * 16 + (lane & 15);   // B/C: N striped over lane&15
    const int koff = (lane >> 4) ? 2 : 0;        // lane half selects K {0,1} vs {2,3}

    v8f c = {};
    #pragma unroll 4
    for (int k0 = 0; k0 < IN_F; k0 += 4) {
        v2f a, b;
        a.x = x[row * IN_F + k0 + koff];
        a.y = x[row * IN_F + k0 + koff + 1];
        b.x = W[(k0 + koff) * 256 + ncol];
        b.y = W[(k0 + koff + 1) * 256 + ncol];
        c = __builtin_amdgcn_wmma_f32_16x16x4_f32(false, a, false, b,
                                                  (short)0, c, false, false);
    }
    // C layout: lane n<16 -> rows m=0..7, lane n+16 -> rows 8..15 (col = lane&15)
    const int mbase = mtile * 16 + ((lane >> 4) ? 8 : 0);
    #pragma unroll
    for (int r = 0; r < 8; ++r) {
        const int m = mbase + r;
        h[m * 256 + ncol] = c[r];
        hT[(size_t)ncol * N_NODES + m] = (_Float16)c[r];
    }
}

// ---------------------------------------------------------------------------
// Kernel 2: el[h][n] = sum_f h[n,h,f]*a[f],  er[h][n] = sum_f h[n,h,f]*a[64+f]
// ---------------------------------------------------------------------------
__global__ void k_el_er(const float* __restrict__ h, const float* __restrict__ a,
                        float* __restrict__ elT, float* __restrict__ erT)
{
    const int t = blockIdx.x * blockDim.x + threadIdx.x;   // 16384 threads
    const int hd = t >> 12;
    const int n  = t & (N_NODES - 1);
    const float* hp = h + n * 256 + hd * F_OUT;
    float sl = 0.f, sr = 0.f;
    #pragma unroll 8
    for (int f = 0; f < F_OUT; ++f) {
        const float v = hp[f];
        sl += v * a[f];
        sr += v * a[F_OUT + f];
    }
    elT[hd * N_NODES + n] = sl;
    erT[hd * N_NODES + n] = sr;
}

// ---------------------------------------------------------------------------
// Kernel 3: fused masked softmax + aggregation.
// Block = 16 output rows, 4 waves (one per head).
// Phase 1 (whole block): stream the 16 adj rows ONCE (branchless): build LDS
//   bitmask + select-max of er per (head,row) -> m = leakyrelu(el + max er),
//   exact because leaky_relu is monotonic.
// Phase 2 (per wave/head): stream j in chunks of 32; build the A-fragment
//   (probabilities, f16, contiguous K = khalf*16 + q, same as B) in registers,
//   accumulate with v_wmma_f32_16x16x32_f16 against hT tiles (B layout:
//   lanes 0-15 hold K 0..15, lanes 16-31 hold K 16..31, contiguous per lane).
// ---------------------------------------------------------------------------
__global__ __launch_bounds__(128) void k_attn(const float* __restrict__ adj,
                                              const float* __restrict__ elT,
                                              const float* __restrict__ erT,
                                              const _Float16* __restrict__ hT,
                                              float* __restrict__ out)
{
    __shared__ unsigned s_mask[16][128];          // 16 rows x 4096 bits
    __shared__ float s_el[H_HEADS][16];
    __shared__ float s_part[H_HEADS][16][8];
    __shared__ float s_m[H_HEADS][16];
    __shared__ float s_sum[H_HEADS][16];
    __shared__ float s_head[H_HEADS][16][F_OUT];  // per-head normalized tiles

    const int tid   = threadIdx.x;
    const int ibase = blockIdx.x * 16;

    if (tid < 64) {
        const int hh = tid >> 4, r = tid & 15;
        s_el[hh][r] = elT[hh * N_NODES + ibase + r];
    }

    // ---- Phase 1: bitmask + branchless masked er max (adj read exactly once) ----
    {
        const int r = tid & 15, slice = tid >> 4;           // 8 slices x 512 j
        const float* arow = adj + (size_t)(ibase + r) * N_NODES + slice * 512;
        float mx0 = -1e30f, mx1 = -1e30f, mx2 = -1e30f, mx3 = -1e30f;
        for (int w = 0; w < 16; ++w) {
            unsigned bits = 0;
            const int jb = slice * 512 + w * 32;
            __builtin_prefetch(arow + w * 32 + 128, 0, 0);
            #pragma unroll
            for (int b4 = 0; b4 < 8; ++b4) {
                const v4f av = *(const v4f*)(arow + w * 32 + b4 * 4);
                const v4f e0 = *(const v4f*)(erT + 0 * N_NODES + jb + b4 * 4);
                const v4f e1 = *(const v4f*)(erT + 1 * N_NODES + jb + b4 * 4);
                const v4f e2 = *(const v4f*)(erT + 2 * N_NODES + jb + b4 * 4);
                const v4f e3 = *(const v4f*)(erT + 3 * N_NODES + jb + b4 * 4);
                #pragma unroll
                for (int cc = 0; cc < 4; ++cc) {
                    const bool on = (av[cc] != 0.f);
                    bits |= (on ? 1u : 0u) << (b4 * 4 + cc);
                    mx0 = fmaxf(mx0, on ? e0[cc] : -1e30f);
                    mx1 = fmaxf(mx1, on ? e1[cc] : -1e30f);
                    mx2 = fmaxf(mx2, on ? e2[cc] : -1e30f);
                    mx3 = fmaxf(mx3, on ? e3[cc] : -1e30f);
                }
            }
            s_mask[r][slice * 16 + w] = bits;
        }
        s_part[0][r][slice] = mx0; s_part[1][r][slice] = mx1;
        s_part[2][r][slice] = mx2; s_part[3][r][slice] = mx3;
    }
    __syncthreads();
    if (tid < 64) {
        const int hh = tid >> 4, r = tid & 15;
        float mx = -1e30f;
        #pragma unroll
        for (int s = 0; s < 8; ++s) mx = fmaxf(mx, s_part[hh][r][s]);
        s_m[hh][r] = leaky(s_el[hh][r] + mx);   // exact row max of masked logits
    }
    __syncthreads();

    // ---- Phase 2: per-head streaming softmax + WMMA aggregation ----
    const int hh    = tid >> 5;            // wave == head
    const int lane  = tid & 31;
    const int r     = lane & 15;           // A-matrix row / B-matrix column
    const int khalf = lane >> 4;
    const float el_r = s_el[hh][r];
    const float m_r  = s_m[hh][r];
    const float* er  = erT + hh * N_NODES;
    const _Float16* hTh = hT + (size_t)hh * F_OUT * N_NODES;

    v8f c0 = {}, c1 = {}, c2 = {}, c3 = {};
    float sp = 0.f;

    for (int jb = 0; jb < N_NODES; jb += 32) {
        const unsigned mask = s_mask[r][jb >> 5];
        const int j0 = jb + khalf * 16;         // A element 0..7  -> K khalf*16+q
        const int j1 = jb + khalf * 16 + 8;     // A element 8..15 -> K khalf*16+q
        v4f eg0 = *(const v4f*)(er + j0);
        v4f eg1 = *(const v4f*)(er + j0 + 4);
        v4f eg2 = *(const v4f*)(er + j1);
        v4f eg3 = *(const v4f*)(er + j1 + 4);

        v16h afrag;
        float sloc = 0.f;
        #pragma unroll
        for (int q = 0; q < 16; ++q) {
            const int bit = khalf * 16 + q;
            const float ev = (q < 4)  ? eg0[q]
                           : (q < 8)  ? eg1[q - 4]
                           : (q < 12) ? eg2[q - 8]
                           :            eg3[q - 12];
            const float e = leaky(el_r + ev) - m_r;
            const float p = ((mask >> bit) & 1u) ? __expf(e) : 0.f;
            afrag[q] = (_Float16)p;
            sloc += p;
        }
        sp += sloc;

        // B fragments: column = r, K = khalf*16 + 0..15 contiguous
        const size_t boff = (size_t)jb + khalf * 16;
        v16h b;
        b  = *(const v16h*)(hTh + (size_t)( 0 + r) * N_NODES + boff);
        c0 = __builtin_amdgcn_wmma_f32_16x16x32_f16(false, afrag, false, b, (short)0, c0, false, false);
        b  = *(const v16h*)(hTh + (size_t)(16 + r) * N_NODES + boff);
        c1 = __builtin_amdgcn_wmma_f32_16x16x32_f16(false, afrag, false, b, (short)0, c1, false, false);
        b  = *(const v16h*)(hTh + (size_t)(32 + r) * N_NODES + boff);
        c2 = __builtin_amdgcn_wmma_f32_16x16x32_f16(false, afrag, false, b, (short)0, c2, false, false);
        b  = *(const v16h*)(hTh + (size_t)(48 + r) * N_NODES + boff);
        c3 = __builtin_amdgcn_wmma_f32_16x16x32_f16(false, afrag, false, b, (short)0, c3, false, false);
    }

    // softmax denominator: combine the two lane halves of each row
    const float rsum = sp + __shfl_xor(sp, 16, 32);
    if (lane < 16) s_sum[hh][lane] = rsum;   // same-wave LDS RAW: in-order DS ops

    // C layout: lane n<16 -> rows 0..7, lane n+16 -> rows 8..15; column = r
    const int mb = khalf * 8;
    #pragma unroll
    for (int q = 0; q < 8; ++q) {
        const float inv = 0.25f / s_sum[hh][mb + q];
        s_head[hh][mb + q][ 0 + r] = c0[q] * inv;
        s_head[hh][mb + q][16 + r] = c1[q] * inv;
        s_head[hh][mb + q][32 + r] = c2[q] * inv;
        s_head[hh][mb + q][48 + r] = c3[q] * inv;
    }
    __syncthreads();

    // mean over heads -> out (16 x 64 per block)
    #pragma unroll
    for (int q = 0; q < 8; ++q) {
        const int idx = tid * 8 + q;            // 0..1023
        const int rr = idx >> 6, f = idx & 63;
        out[(size_t)(ibase + rr) * F_OUT + f] =
            s_head[0][rr][f] + s_head[1][rr][f] + s_head[2][rr][f] + s_head[3][rr][f];
    }
}

// ---------------------------------------------------------------------------
extern "C" void kernel_launch(void* const* d_in, const int* in_sizes, int n_in,
                              void* d_out, int out_size, void* d_ws, size_t ws_size,
                              hipStream_t stream)
{
    const float* x   = (const float*)d_in[0];   // (4096, 256)
    const float* adj = (const float*)d_in[1];   // (4096, 4096)
    const float* W   = (const float*)d_in[2];   // (256, 256)
    const float* a   = (const float*)d_in[3];   // (128, 1)
    float* out = (float*)d_out;                 // (4096, 64)

    char* ws = (char*)d_ws;
    float*    h   = (float*)(ws);                                  // 4 MB
    _Float16* hT  = (_Float16*)(ws + (size_t)4 * 1024 * 1024);     // 2 MB
    float*    elT = (float*)(ws + (size_t)6 * 1024 * 1024);        // 64 KB
    float*    erT = (float*)(ws + (size_t)6 * 1024 * 1024 + 65536);// 64 KB

    k_gemm_xw<<<1024, 128, 0, stream>>>(x, W, h, hT);
    k_el_er <<<64, 256, 0, stream>>>(h, a, elT, erT);
    k_attn  <<<256, 128, 0, stream>>>(adj, elT, erT, hT, out);
}
